// HyperFast_57509612093866
// MI455X (gfx1250) — compile-verified, matrix operand order checked by reading, not run on hardware
//
#include <hip/hip_runtime.h>
#include <hip/hip_bf16.h>

// ---------------------------------------------------------------------------
// HyperFast forward on MI455X (gfx1250, wave32, WMMA).
// All GEMMs use v_wmma_f32_16x16x32_bf16 with 128x128 block tiles staged in
// LDS via GLOBAL_LOAD_ASYNC_TO_LDS_B128 (ASYNCcnt-tracked, double-buffered)
// when the toolchain exposes the builtin; sync b128 copies otherwise.
// SVD is replaced by covariance + GEMM-only subspace iteration with
// Newton-Schulz orthonormalization so the whole pipeline rides the WMMA path.
// NOTE: the GEMM requires M % 128 == 0 (all call sites comply); B operand
// rows are padded to a multiple of 128 by the converters.
// ---------------------------------------------------------------------------

typedef __bf16 bf16_t;
typedef __attribute__((ext_vector_type(16))) __bf16 v16bf;
typedef __attribute__((ext_vector_type(8)))  float  v8f;
typedef int v4i_t __attribute__((vector_size(16)));

#if defined(__AMDGCN__) && \
    __has_builtin(__builtin_amdgcn_global_load_async_to_lds_b128) && \
    __has_builtin(__builtin_amdgcn_s_wait_asynccnt)
#define ASYNC_LDS 1
typedef __attribute__((address_space(1))) v4i_t* as1_v4i;
typedef __attribute__((address_space(3))) v4i_t* as3_v4i;
#define WAIT_ASYNC(n) __builtin_amdgcn_s_wait_asynccnt(n)
#else
#define ASYNC_LDS 0
#define WAIT_ASYNC(n) ((void)0)
#endif

#define NROW   4096
#define DIN    784
#define KPADX  800     // 784 padded to mult of 32
#define RFN    1024
#define HIDN   1024
#define ND     256
#define D0     814     // 3*256+46
#define D0P    832
#define D1     1070    // 4*256+46
#define D1P    1088
#define OWN    65792   // 257*256
#define NCLS   10
#define MAXCAT 46
#define CLIPV  27.6041f

// ===========================================================================
// WMMA GEMM:  C(M,N) = act( A(MxK,bf16) * B(NxK,bf16)^T + bias + res )
// Block tile 128x128, 8 waves (4x2), each wave 32x64 = 2x4 WMMA tiles.
// Double-buffered LDS; async fills when available.
// ===========================================================================
#define BM 128
#define BN 128
#define BK 32
#define LDSS 40   // bf16 row stride in LDS (32 + 8 pad -> 80B, 16B aligned)

union FragU { uint4 q[2]; v16bf v; };

__device__ __forceinline__ void tile_ld16(const bf16_t* __restrict__ g, bf16_t* l)
{
#if ASYNC_LDS
    __builtin_amdgcn_global_load_async_to_lds_b128((as1_v4i)(g), (as3_v4i)(l), 0, 0);
#else
    *reinterpret_cast<uint4*>(l) = *reinterpret_cast<const uint4*>(g);
#endif
}

__global__ __launch_bounds__(256)
void k_gemm_nt(const bf16_t* __restrict__ A, const bf16_t* __restrict__ B,
               const float* __restrict__ bias, const float* __restrict__ res,
               float* __restrict__ Cf, bf16_t* __restrict__ Cb,
               int M, int N, int K, int lda, int ldb, int relu)
{
    __shared__ bf16_t As[2][BM * LDSS];
    __shared__ bf16_t Bs[2][BN * LDSS];

    const int tid  = threadIdx.x;
    const int lane = tid & 31;
    const int wid  = tid >> 5;
    const int wm   = wid & 3;          // 0..3  (M direction)
    const int wn   = wid >> 2;         // 0..1  (N direction)
    const int m0   = blockIdx.y * BM;
    const int n0   = blockIdx.x * BN;

    v8f acc[2][4];
#pragma unroll
    for (int i = 0; i < 2; ++i)
#pragma unroll
        for (int j = 0; j < 4; ++j)
            acc[i][j] = (v8f){0.f,0.f,0.f,0.f,0.f,0.f,0.f,0.f};

    const int lr = tid >> 2;          // 0..63 : tile row
    const int lc = (tid & 3) * 8;     // element offset 0/8/16/24 (16B chunks)

    const int koff = (lane >> 4) * 8; // CDNA5 16-bit A/B frag K sub-offset
    const int fr   = lane & 15;       // M (A) / N (B) index within 16-tile

    // issue one block tile (4 x b128 per thread) into LDS buffer `buf`
    auto issue = [&](int kb, int buf) {
        tile_ld16(A + (size_t)(m0 + lr)      * lda + kb + lc, &As[buf][lr * LDSS + lc]);
        tile_ld16(A + (size_t)(m0 + lr + 64) * lda + kb + lc, &As[buf][(lr + 64) * LDSS + lc]);
        tile_ld16(B + (size_t)(n0 + lr)      * ldb + kb + lc, &Bs[buf][lr * LDSS + lc]);
        tile_ld16(B + (size_t)(n0 + lr + 64) * ldb + kb + lc, &Bs[buf][(lr + 64) * LDSS + lc]);
    };

    const int nsteps = K / BK;
    issue(0, 0);

    for (int s = 0; s < nsteps; ++s) {
        const int buf = s & 1;
        if (s + 1 < nsteps) {
            issue((s + 1) * BK, buf ^ 1);   // overlap next fill with compute
            WAIT_ASYNC(4);                  // tile `s` (4 issues) retired
        } else {
            WAIT_ASYNC(0);
        }
        __syncthreads();

        // ---- fragments from LDS per documented CDNA5 16-bit layouts ----
        v16bf afrag[2], bfrag[4];
#pragma unroll
        for (int mi = 0; mi < 2; ++mi) {
            int r = wm * 32 + mi * 16 + fr;
            FragU u;
            u.q[0] = *reinterpret_cast<const uint4*>(&As[buf][r * LDSS + koff]);
            u.q[1] = *reinterpret_cast<const uint4*>(&As[buf][r * LDSS + 16 + koff]);
            afrag[mi] = u.v;
        }
#pragma unroll
        for (int nj = 0; nj < 4; ++nj) {
            int r = wn * 64 + nj * 16 + fr;
            FragU u;
            u.q[0] = *reinterpret_cast<const uint4*>(&Bs[buf][r * LDSS + koff]);
            u.q[1] = *reinterpret_cast<const uint4*>(&Bs[buf][r * LDSS + 16 + koff]);
            bfrag[nj] = u.v;
        }

#pragma unroll
        for (int mi = 0; mi < 2; ++mi)
#pragma unroll
            for (int nj = 0; nj < 4; ++nj)
                acc[mi][nj] = __builtin_amdgcn_wmma_f32_16x16x32_bf16(
                    false, afrag[mi], false, bfrag[nj],
                    (short)0, acc[mi][nj], false, false);
        __syncthreads();   // protect buf before it is refilled next step
    }

    // ---- epilogue (C layout: lane&15 = N, lane>=16 -> M += 8, vgpr e = M) ----
    const int colb = n0 + wn * 64 + fr;
    const int mhi  = (lane >> 4) * 8;
#pragma unroll
    for (int mi = 0; mi < 2; ++mi) {
#pragma unroll
        for (int e = 0; e < 8; ++e) {
            int row = m0 + wm * 32 + mi * 16 + mhi + e;
#pragma unroll
            for (int nj = 0; nj < 4; ++nj) {
                int col = colb + nj * 16;
                if (col >= N) continue;
                float x = acc[mi][nj][e];
                if (bias) x += bias[col];
                if (res)  x += res[(size_t)row * N + col];
                if (relu) x = fmaxf(x, 0.f);
                if (Cf) Cf[(size_t)row * N + col] = x;
                if (Cb) Cb[(size_t)row * N + col] = (bf16_t)x;
            }
        }
    }
}

// ===========================================================================
// Helper kernels
// ===========================================================================

// f32 -> bf16 with zero padding (rows x cols  ->  rowsP x colsP)
__global__ void k_cvt_pad(const float* __restrict__ src, bf16_t* __restrict__ dst,
                          int rows, int cols, int rowsP, int colsP)
{
    int idx = blockIdx.x * blockDim.x + threadIdx.x;
    int tot = rowsP * colsP;
    if (idx >= tot) return;
    int r = idx / colsP, c = idx % colsP;
    float v = (r < rows && c < cols) ? src[(size_t)r * cols + c] : 0.f;
    dst[idx] = (bf16_t)v;
}

template <typename TS, typename TD>
__global__ void k_transpose(const TS* __restrict__ src, TD* __restrict__ dst, int R, int C)
{
    __shared__ TS tile[32][33];
    int bx = blockIdx.x * 32, by = blockIdx.y * 32;
    int x = bx + threadIdx.x;
    for (int j = threadIdx.y; j < 32; j += 8) {
        int yy = by + j;
        if (yy < R && x < C) tile[j][threadIdx.x] = src[(size_t)yy * C + x];
    }
    __syncthreads();
    x = by + threadIdx.x;
    for (int j = threadIdx.y; j < 32; j += 8) {
        int yy = bx + j;
        if (yy < C && x < R) dst[(size_t)yy * R + x] = (TD)tile[threadIdx.x][j];
    }
}

__global__ void k_colmean_f32(const float* __restrict__ x, int M, int D, int ld,
                              float* __restrict__ mu)
{
    int c = blockIdx.x * blockDim.x + threadIdx.x;
    if (c >= D) return;
    float s = 0.f;
    for (int r = 0; r < M; ++r) s += x[(size_t)r * ld + c];
    mu[c] = s / (float)M;
}

__global__ void k_colmean_bf16(const bf16_t* __restrict__ x, int M, int D, int ld,
                               float* __restrict__ mu)
{
    int c = blockIdx.x * blockDim.x + threadIdx.x;
    if (c >= D) return;
    float s = 0.f;
    for (int r = 0; r < M; ++r) s += (float)x[(size_t)r * ld + c];
    mu[c] = s / (float)M;
}

__global__ void k_center_cvt(const float* __restrict__ Xr, const float* __restrict__ mu,
                             bf16_t* __restrict__ dst, int M, int D)
{
    int idx = blockIdx.x * blockDim.x + threadIdx.x;
    if (idx >= M * D) return;
    int c = idx % D;
    dst[idx] = (bf16_t)(Xr[idx] - mu[c]);
}

__global__ void k_init_V(bf16_t* __restrict__ V, int n)
{
    int idx = blockIdx.x * blockDim.x + threadIdx.x;
    if (idx >= n) return;
    float v = sinf((float)idx * 12.9898f) * 43758.547f;
    v = v - truncf(v);
    V[idx] = (bf16_t)(v - 0.5f);
}

__global__ void k_sumsq(const float* __restrict__ x, int n, float* __restrict__ out)
{
    __shared__ float s[256];
    float a = 0.f;
    for (int i = blockIdx.x * 256 + threadIdx.x; i < n; i += gridDim.x * 256)
        a += x[i] * x[i];
    s[threadIdx.x] = a;
    __syncthreads();
    for (int o = 128; o > 0; o >>= 1) {
        if (threadIdx.x < o) s[threadIdx.x] += s[threadIdx.x + o];
        __syncthreads();
    }
    if (threadIdx.x == 0) atomicAdd(out, s[0]);
}

__global__ void k_scale_cvt(const float* __restrict__ src, bf16_t* __restrict__ dst,
                            int n, const float* __restrict__ sumsq)
{
    int idx = blockIdx.x * blockDim.x + threadIdx.x;
    if (idx >= n) return;
    float sc = rsqrtf(*sumsq + 1e-20f);
    dst[idx] = (bf16_t)(src[idx] * sc);
}

// H = 1.5*I - 0.5*G   (256x256, f32 -> bf16)
__global__ void k_ns_h(const float* __restrict__ G, bf16_t* __restrict__ H)
{
    int idx = blockIdx.x * blockDim.x + threadIdx.x;
    if (idx >= ND * ND) return;
    int i = idx / ND, j = idx % ND;
    float v = (i == j ? 1.5f : 0.f) - 0.5f * G[idx];
    H[idx] = (bf16_t)v;
}

// per-column sign flip (by max-|.| element) + clip; writes f32 inplace + bf16
__global__ void k_signflip_clip(float* __restrict__ out, bf16_t* __restrict__ outb,
                                int M, int D)
{
    __shared__ float sa[256], sv[256];
    int j = blockIdx.x, t = threadIdx.x;
    float ba = -1.f, bv = 0.f;
    for (int r = t; r < M; r += 256) {
        float v = out[(size_t)r * D + j];
        float a = fabsf(v);
        if (a > ba) { ba = a; bv = v; }
    }
    sa[t] = ba; sv[t] = bv;
    __syncthreads();
    for (int s = 128; s > 0; s >>= 1) {
        if (t < s && sa[t + s] > sa[t]) { sa[t] = sa[t + s]; sv[t] = sv[t + s]; }
        __syncthreads();
    }
    float sgn = (sv[0] >= 0.f) ? 1.f : -1.f;
    for (int r = t; r < M; r += 256) {
        float v = out[(size_t)r * D + j] * sgn;
        v = fminf(fmaxf(v, -CLIPV), CLIPV);
        out[(size_t)r * D + j]  = v;
        outb[(size_t)r * D + j] = (bf16_t)v;
    }
}

__global__ void k_counts(const int* __restrict__ y, float* __restrict__ counts, int n)
{
    int i = blockIdx.x * blockDim.x + threadIdx.x;
    if (i < n) atomicAdd(&counts[y[i]], 1.f);
}

__global__ void k_class_acc(const float* __restrict__ v, const int* __restrict__ y,
                            float* __restrict__ csum, int M, int D, int ld)
{
    int idx = blockIdx.x * blockDim.x + threadIdx.x;
    if (idx >= M * D) return;
    int r = idx / D, d = idx % D;
    atomicAdd(&csum[y[r] * D + d], v[(size_t)r * ld + d]);
}

__global__ void k_class_final(const float* __restrict__ csum, const float* __restrict__ counts,
                              const float* __restrict__ fb, float* __restrict__ cmean, int D)
{
    int idx = blockIdx.x * blockDim.x + threadIdx.x;
    if (idx >= NCLS * D) return;
    int c = idx / D, d = idx % D;
    float n = counts[c];
    cmean[idx] = (n > 0.5f) ? csum[idx] / n : fb[d];
}

// data0 = [out | gmean | cmean[y] | onehot | pad]  -> bf16 (NROW x D0P)
__global__ void k_build_data0(const float* __restrict__ out, const float* __restrict__ gmean,
                              const float* __restrict__ cmean, const int* __restrict__ y,
                              bf16_t* __restrict__ dst)
{
    int idx = blockIdx.x * blockDim.x + threadIdx.x;
    if (idx >= NROW * D0P) return;
    int r = idx / D0P, c = idx % D0P;
    float v;
    if (c < 256)       v = out[(size_t)r * 256 + c];
    else if (c < 512)  v = gmean[c - 256];
    else if (c < 768)  v = cmean[y[r] * 256 + (c - 512)];
    else if (c < D0)   v = (y[r] == (c - 768)) ? 1.f : 0.f;
    else               v = 0.f;
    dst[idx] = (bf16_t)v;
}

// data1 = [newout | pca(from data0 cols 0..767) | onehot | pad] -> bf16 (NROW x D1P)
__global__ void k_build_data1(const float* __restrict__ newout, const bf16_t* __restrict__ data0,
                              const int* __restrict__ y, bf16_t* __restrict__ dst)
{
    int idx = blockIdx.x * blockDim.x + threadIdx.x;
    if (idx >= NROW * D1P) return;
    int r = idx / D1P, c = idx % D1P;
    if (c < 256)       dst[idx] = (bf16_t)newout[(size_t)r * 256 + c];
    else if (c < 1024) dst[idx] = data0[(size_t)r * D0P + (c - 256)];
    else if (c < D1)   dst[idx] = (bf16_t)((y[r] == (c - 1024)) ? 1.f : 0.f);
    else               dst[idx] = (bf16_t)0.f;
}

// wvec[r] = dot(W[r,:], v) + b[r]   (one wave per row; f32, bandwidth bound)
__global__ __launch_bounds__(256)
void k_matvec(const float* __restrict__ W, const float* __restrict__ b,
              const float* __restrict__ v, float* __restrict__ out, int R, int K)
{
    __shared__ float sv[HIDN];
    for (int i = threadIdx.x; i < K; i += 256) sv[i] = v[i];
    __syncthreads();
    int lane = threadIdx.x & 31, wid = threadIdx.x >> 5;
    int row = blockIdx.x * 8 + wid;
    if (row >= R) return;
    const float4* wr = reinterpret_cast<const float4*>(W + (size_t)row * K);
    float s = 0.f;
    for (int i = 0; i < K / 128; ++i) {
        int q = i * 32 + lane;
        float4 w4 = wr[q];
        s += w4.x * sv[q * 4 + 0] + w4.y * sv[q * 4 + 1] +
             w4.z * sv[q * 4 + 2] + w4.w * sv[q * 4 + 3];
    }
    for (int o = 16; o > 0; o >>= 1) s += __shfl_down(s, o, 32);
    if (lane == 0) out[row] = s + b[row];
}

// wvec (257*256) -> wT bf16 (256x256, transposed) + bias f32 (256)
__global__ void k_split_w(const float* __restrict__ wvec, bf16_t* __restrict__ wT,
                          float* __restrict__ wbias)
{
    int idx = blockIdx.x * blockDim.x + threadIdx.x;
    if (idx < 65536) {
        int n = idx / 256, k = idx % 256;
        wT[idx] = (bf16_t)wvec[k * 256 + n];
    } else if (idx < OWN) {
        wbias[idx - 65536] = wvec[idx];
    }
}

// Wc (10 x 257) = [wmean[:, :256] + imean | wmean[:, 256]]
__global__ void k_build_Wc(const float* __restrict__ wmean, const float* __restrict__ imean,
                           float* __restrict__ Wc)
{
    int idx = blockIdx.x * blockDim.x + threadIdx.x;
    if (idx >= NCLS * 257) return;
    int c = idx / 257, d = idx % 257;
    Wc[idx] = (d < 256) ? (wmean[c * 257 + d] + imean[c * 256 + d])
                        : wmean[c * 257 + 256];
}

__global__ __launch_bounds__(256)
void k_logits(const float* __restrict__ out2, const float* __restrict__ Wc,
              float* __restrict__ dout, int M)
{
    __shared__ float sW[NCLS * 257];
    for (int i = threadIdx.x; i < NCLS * 257; i += 256) sW[i] = Wc[i];
    __syncthreads();
    int i = blockIdx.x * 256 + threadIdx.x;
    if (i >= M) return;
    float acc[NCLS];
#pragma unroll
    for (int c = 0; c < NCLS; ++c) acc[c] = sW[c * 257 + 256];
    for (int k = 0; k < 256; ++k) {
        float x = out2[(size_t)i * 256 + k];
#pragma unroll
        for (int c = 0; c < NCLS; ++c) acc[c] += x * sW[c * 257 + k];
    }
#pragma unroll
    for (int c = 0; c < NCLS; ++c) dout[i * NCLS + c] = acc[c];
}

// ===========================================================================
// Host orchestration
// ===========================================================================
static inline void gemm(hipStream_t s, const bf16_t* A, const bf16_t* B,
                        const float* bias, const float* res,
                        float* Cf, bf16_t* Cb, int M, int N, int K,
                        int lda, int ldb, int relu)
{
    dim3 g((N + BN - 1) / BN, (M + BM - 1) / BM), b(256);
    k_gemm_nt<<<g, b, 0, s>>>(A, B, bias, res, Cf, Cb, M, N, K, lda, ldb, relu);
}

#define GRID1(n) dim3(((n) + 255) / 256), dim3(256)

extern "C" void kernel_launch(void* const* d_in, const int* in_sizes, int n_in,
                              void* d_out, int out_size, void* d_ws, size_t ws_size,
                              hipStream_t stream)
{
    (void)in_sizes; (void)n_in; (void)out_size; (void)ws_size;
    const float* X      = (const float*)d_in[0];
    const int*   y      = (const int*)  d_in[1];
    const float* rf_w   = (const float*)d_in[3];
    const float* rf_b   = (const float*)d_in[4];
    const float* w_in0  = (const float*)d_in[5];
    const float* b_in0  = (const float*)d_in[6];
    const float* w_in1  = (const float*)d_in[7];
    const float* b_in1  = (const float*)d_in[8];
    const float* mw1    = (const float*)d_in[9];
    const float* mb1    = (const float*)d_in[10];
    const float* mw2    = (const float*)d_in[11];
    const float* mb2    = (const float*)d_in[12];
    const float* ew0    = (const float*)d_in[13];
    const float* eb0    = (const float*)d_in[14];
    const float* ew1    = (const float*)d_in[15];
    const float* eb1    = (const float*)d_in[16];
    const float* w_inL  = (const float*)d_in[17];
    const float* b_inL  = (const float*)d_in[18];
    const float* w_outL = (const float*)d_in[19];
    const float* b_outL = (const float*)d_in[20];
    float* dout = (float*)d_out;

    // ---- workspace carve-out ----
    char*  base = (char*)d_ws;
    size_t off  = 0;
    auto alloc = [&](size_t bytes) -> void* {
        void* p = base + off;
        off = (off + bytes + 255) & ~(size_t)255;
        return p;
    };
    bf16_t* Xbf    = (bf16_t*)alloc((size_t)NROW * KPADX * 2);
    bf16_t* rfwbf  = (bf16_t*)alloc((size_t)RFN * KPADX * 2);
    float*  Xr     = (float*) alloc((size_t)NROW * RFN * 4);
    float*  mu     = (float*) alloc(RFN * 4);
    bf16_t* Xcbf   = (bf16_t*)alloc((size_t)NROW * RFN * 2);
    bf16_t* XcTbf  = (bf16_t*)alloc((size_t)RFN * NROW * 2);
    float*  Cf     = (float*) alloc((size_t)RFN * RFN * 4);
    bf16_t* Cbf    = (bf16_t*)alloc((size_t)RFN * RFN * 2);
    bf16_t* Ybf    = (bf16_t*)alloc((size_t)ND * RFN * 2);    // Vt (256x1024)
    float*  Tm     = (float*) alloc((size_t)RFN * ND * 4);
    float*  Yf     = (float*) alloc((size_t)ND * RFN * 4);
    bf16_t* Yt2    = (bf16_t*)alloc((size_t)RFN * ND * 2);
    float*  Gf     = (float*) alloc((size_t)ND * ND * 4);
    bf16_t* Hbf    = (bf16_t*)alloc((size_t)ND * ND * 2);
    float*  sumsq  = (float*) alloc(256);
    float*  outf   = (float*) alloc((size_t)NROW * ND * 4);
    bf16_t* outbf  = (bf16_t*)alloc((size_t)NROW * ND * 2);
    float*  gmean  = (float*) alloc(ND * 4);
    float*  counts = (float*) alloc(NCLS * 4);
    float*  csum   = (float*) alloc(NCLS * 260 * 4);
    float*  cmean  = (float*) alloc(NCLS * ND * 4);
    bf16_t* data0  = (bf16_t*)alloc((size_t)NROW * D0P * 2);
    bf16_t* data1  = (bf16_t*)alloc((size_t)NROW * D1P * 2);
    bf16_t* win0b  = (bf16_t*)alloc((size_t)HIDN * D0P * 2);
    bf16_t* win1b  = (bf16_t*)alloc((size_t)HIDN * D1P * 2);
    bf16_t* winLb  = (bf16_t*)alloc((size_t)HIDN * D1P * 2);
    bf16_t* mw1b   = (bf16_t*)alloc((size_t)HIDN * HIDN * 2);
    bf16_t* mw2b   = (bf16_t*)alloc((size_t)HIDN * HIDN * 2);
    bf16_t* woutLb = (bf16_t*)alloc((size_t)384 * HIDN * 2);  // 257 rows -> 384
    bf16_t* hA     = (bf16_t*)alloc((size_t)NROW * HIDN * 2);
    bf16_t* hB     = (bf16_t*)alloc((size_t)NROW * HIDN * 2);
    float*  hmean  = (float*) alloc(HIDN * 4);
    float*  wvec   = (float*) alloc((size_t)OWN * 4);
    bf16_t* wTbf   = (bf16_t*)alloc((size_t)ND * ND * 2);
    float*  wbias  = (float*) alloc(ND * 4);
    float*  out1f  = (float*) alloc((size_t)NROW * ND * 4);
    bf16_t* out1bf = (bf16_t*)alloc((size_t)NROW * ND * 2);
    float*  out2f  = (float*) alloc((size_t)NROW * ND * 4);
    bf16_t* out2bf = (bf16_t*)alloc((size_t)NROW * ND * 2);
    float*  wps    = (float*) alloc((size_t)NROW * 257 * 4);
    float*  csum2  = (float*) alloc(NCLS * 260 * 4);
    float*  wmean  = (float*) alloc(NCLS * 257 * 4);
    float*  imean  = (float*) alloc(NCLS * ND * 4);
    float*  wpsmu  = (float*) alloc(257 * 4);
    float*  o2mu   = (float*) alloc(ND * 4);
    float*  Wc     = (float*) alloc(NCLS * 257 * 4);

    dim3 tB(32, 8);

    // ---- 0) weight / input conversions (f32 -> padded bf16) ----
    k_cvt_pad<<<GRID1(NROW * KPADX), 0, stream>>>(X,      Xbf,   NROW, DIN, NROW, KPADX);
    k_cvt_pad<<<GRID1(RFN  * KPADX), 0, stream>>>(rf_w,   rfwbf, RFN,  DIN, RFN,  KPADX);
    k_cvt_pad<<<GRID1(HIDN * D0P),   0, stream>>>(w_in0,  win0b, HIDN, D0,  HIDN, D0P);
    k_cvt_pad<<<GRID1(HIDN * D1P),   0, stream>>>(w_in1,  win1b, HIDN, D1,  HIDN, D1P);
    k_cvt_pad<<<GRID1(HIDN * D1P),   0, stream>>>(w_inL,  winLb, HIDN, D1,  HIDN, D1P);
    k_cvt_pad<<<GRID1(HIDN * HIDN),  0, stream>>>(mw1,    mw1b,  HIDN, HIDN, HIDN, HIDN);
    k_cvt_pad<<<GRID1(HIDN * HIDN),  0, stream>>>(mw2,    mw2b,  HIDN, HIDN, HIDN, HIDN);
    k_cvt_pad<<<GRID1(384  * HIDN),  0, stream>>>(w_outL, woutLb, 257, HIDN, 384, HIDN);

    // ---- 1) random features: Xr = relu(X @ rf_w^T + rf_b) ----
    gemm(stream, Xbf, rfwbf, rf_b, nullptr, Xr, nullptr, NROW, RFN, KPADX, KPADX, KPADX, 1);

    // ---- 2) center + bf16, and transpose for covariance ----
    k_colmean_f32<<<GRID1(RFN), 0, stream>>>(Xr, NROW, RFN, RFN, mu);
    k_center_cvt<<<GRID1(NROW * RFN), 0, stream>>>(Xr, mu, Xcbf, NROW, RFN);
    k_transpose<bf16_t, bf16_t><<<dim3(RFN / 32, NROW / 32), tB, 0, stream>>>(Xcbf, XcTbf, NROW, RFN);

    // ---- 3) covariance C = Xc^T Xc ----
    gemm(stream, XcTbf, XcTbf, nullptr, nullptr, Cf, Cbf, RFN, RFN, NROW, NROW, NROW, 0);

    // ---- 4) top-256 subspace via power iteration + Newton-Schulz orth ----
    k_init_V<<<GRID1(ND * RFN), 0, stream>>>(Ybf, ND * RFN);
    for (int it = 0; it < 8; ++it) {
        gemm(stream, Cbf, Ybf, nullptr, nullptr, Tm, nullptr, RFN, ND, RFN, RFN, RFN, 0);
        k_transpose<float, float><<<dim3(ND / 32, RFN / 32), tB, 0, stream>>>(Tm, Yf, RFN, ND);
        (void)hipMemsetAsync(sumsq, 0, 4, stream);
        k_sumsq<<<dim3(256), dim3(256), 0, stream>>>(Yf, ND * RFN, sumsq);
        k_scale_cvt<<<GRID1(ND * RFN), 0, stream>>>(Yf, Ybf, ND * RFN, sumsq);
        for (int ns = 0; ns < 3; ++ns) {
            gemm(stream, Ybf, Ybf, nullptr, nullptr, Gf, nullptr, ND, ND, RFN, RFN, RFN, 0);
            k_ns_h<<<GRID1(ND * ND), 0, stream>>>(Gf, Hbf);
            k_transpose<bf16_t, bf16_t><<<dim3(RFN / 32, ND / 32), tB, 0, stream>>>(Ybf, Yt2, ND, RFN);
            gemm(stream, Hbf, Yt2, nullptr, nullptr, nullptr, Ybf, ND, RFN, ND, ND, ND, 0);
        }
    }

    // ---- 5) out = Xc @ V (= U*S), sign flip, clip ----
    gemm(stream, Xcbf, Ybf, nullptr, nullptr, outf, nullptr, NROW, ND, RFN, RFN, RFN, 0);
    k_signflip_clip<<<dim3(ND), dim3(256), 0, stream>>>(outf, outbf, NROW, ND);

    // ---- 6) stats + data0 ----
    k_colmean_f32<<<GRID1(ND), 0, stream>>>(outf, NROW, ND, ND, gmean);
    (void)hipMemsetAsync(counts, 0, NCLS * 4, stream);
    (void)hipMemsetAsync(csum, 0, NCLS * ND * 4, stream);
    k_counts<<<GRID1(NROW), 0, stream>>>(y, counts, NROW);
    k_class_acc<<<GRID1(NROW * ND), 0, stream>>>(outf, y, csum, NROW, ND, ND);
    k_class_final<<<GRID1(NCLS * ND), 0, stream>>>(csum, counts, gmean, cmean, ND);
    k_build_data0<<<GRID1(NROW * D0P), 0, stream>>>(outf, gmean, cmean, y, data0);

    // ---- 7) layer 0: trunk -> hyper weights -> apply ----
    gemm(stream, data0, win0b, b_in0, nullptr, nullptr, hA, NROW, HIDN, D0P, D0P, D0P, 1);
    gemm(stream, hA, mw1b, mb1, nullptr, nullptr, hB, NROW, HIDN, HIDN, HIDN, HIDN, 1);
    gemm(stream, hB, mw2b, mb2, nullptr, nullptr, hA, NROW, HIDN, HIDN, HIDN, HIDN, 1);
    k_colmean_bf16<<<GRID1(HIDN), 0, stream>>>(hA, NROW, HIDN, HIDN, hmean);
    k_matvec<<<dim3(OWN / 8), dim3(256), 0, stream>>>(ew0, eb0, hmean, wvec, OWN, HIDN);
    k_split_w<<<GRID1(OWN), 0, stream>>>(wvec, wTbf, wbias);
    gemm(stream, outbf, wTbf, wbias, nullptr, out1f, out1bf, NROW, ND, ND, ND, ND, 1);

    // ---- 8) layer 1 (residual) ----
    k_build_data1<<<GRID1(NROW * D1P), 0, stream>>>(out1f, data0, y, data1);
    gemm(stream, data1, win1b, b_in1, nullptr, nullptr, hA, NROW, HIDN, D1P, D1P, D1P, 1);
    gemm(stream, hA, mw1b, mb1, nullptr, nullptr, hB, NROW, HIDN, HIDN, HIDN, HIDN, 1);
    gemm(stream, hB, mw2b, mb2, nullptr, nullptr, hA, NROW, HIDN, HIDN, HIDN, HIDN, 1);
    k_colmean_bf16<<<GRID1(HIDN), 0, stream>>>(hA, NROW, HIDN, HIDN, hmean);
    k_matvec<<<dim3(OWN / 8), dim3(256), 0, stream>>>(ew1, eb1, hmean, wvec, OWN, HIDN);
    k_split_w<<<GRID1(OWN), 0, stream>>>(wvec, wTbf, wbias);
    gemm(stream, out1bf, wTbf, wbias, outf, out2f, out2bf, NROW, ND, ND, ND, ND, 1);

    // ---- 9) last layer: per-sample weights ----
    k_build_data1<<<GRID1(NROW * D1P), 0, stream>>>(out2f, data0, y, data1);
    gemm(stream, data1, winLb, b_inL, nullptr, nullptr, hA, NROW, HIDN, D1P, D1P, D1P, 1);
    gemm(stream, hA, mw1b, mb1, nullptr, nullptr, hB, NROW, HIDN, HIDN, HIDN, HIDN, 1);
    gemm(stream, hB, mw2b, mb2, nullptr, nullptr, hA, NROW, HIDN, HIDN, HIDN, HIDN, 1);
    gemm(stream, hA, woutLb, b_outL, nullptr, wps, nullptr, NROW, 257, HIDN, HIDN, HIDN, 0);

    // ---- 10) class means -> classifier -> logits ----
    k_colmean_f32<<<GRID1(257), 0, stream>>>(wps, NROW, 257, 257, wpsmu);
    (void)hipMemsetAsync(csum2, 0, NCLS * 257 * 4, stream);
    k_class_acc<<<GRID1(NROW * 257), 0, stream>>>(wps, y, csum2, NROW, 257, 257);
    k_class_final<<<GRID1(NCLS * 257), 0, stream>>>(csum2, counts, wpsmu, wmean, 257);

    k_colmean_f32<<<GRID1(ND), 0, stream>>>(out2f, NROW, ND, ND, o2mu);
    (void)hipMemsetAsync(csum, 0, NCLS * ND * 4, stream);
    k_class_acc<<<GRID1(NROW * ND), 0, stream>>>(out2f, y, csum, NROW, ND, ND);
    k_class_final<<<GRID1(NCLS * ND), 0, stream>>>(csum, counts, o2mu, imean, ND);

    k_build_Wc<<<GRID1(NCLS * 257), 0, stream>>>(wmean, imean, Wc);
    k_logits<<<dim3(NROW / 256), dim3(256), 0, stream>>>(out2f, Wc, dout, NROW);
}
